// TrajectoryEncoder_32976758899217
// MI455X (gfx1250) — matrix operand order repaired and checked by reading, not compiled, not run on hardware
//
#include <hip/hip_runtime.h>
#include <hip/hip_bf16.h>

typedef _Float16 half_t;
typedef __attribute__((ext_vector_type(16))) _Float16 v16h;
typedef __attribute__((ext_vector_type(4)))  _Float16 v4h;
typedef __attribute__((ext_vector_type(8)))  float    v8f;
typedef __attribute__((ext_vector_type(4)))  unsigned int u32x4;
typedef __attribute__((ext_vector_type(8)))  int      i32x8;
typedef __attribute__((ext_vector_type(4)))  int      i32x4;

#define Bb 16
#define Tt 64
#define Nn 32
#define Dd 512
#define Hh 4
#define HD 128
#define MROWS (Bb*Tt*Nn)   // 32768
#define NGRAPH (Bb*Tt)     // 1024

// ---------------- WMMA helpers (CDNA5 16x16x32 f16 layouts) ----------------

__device__ __forceinline__ v8f wmma_f16(v16h a, v16h b, v8f c) {
  return __builtin_amdgcn_wmma_f32_16x16x32_f16(false, a, false, b, (short)0, c,
                                                false, false);
}

// A fragment 16x32 (MxK): lane<16 -> row=row0+lane, halves 0..7 = K 0..7,
// halves 8..15 = K 16..23 ; lane>=16 -> same row, K-groups shifted by 8.
__device__ __forceinline__ v16h frag_a_lds(const half_t* base, int ld,
                                           int row0, int k0, int lane) {
  int r  = row0 + (lane & 15);
  int kb = k0 + ((lane & 16) ? 8 : 0);
  v16h f;
#pragma unroll
  for (int j = 0; j < 8; ++j) f[j] = base[r * ld + kb + j];
#pragma unroll
  for (int j = 0; j < 8; ++j) f[8 + j] = base[r * ld + kb + 16 + j];
  return f;
}

__device__ __forceinline__ v4h cvt4(float4 f) {
  v4h h;
  h[0] = (half_t)f.x; h[1] = (half_t)f.y; h[2] = (half_t)f.z; h[3] = (half_t)f.w;
  return h;
}

// ---------------- Tensor Data Mover: f16 tile (32 rows x 64 cols) -> LDS ----
// Tensor: 512x512 f16 row-major. Tile rows are 128B; LDS dest row stride is
// 144B, expressed via D# padding (pad_interval = 32 dwords, pad_amount = 4
// dwords). Descriptor per CDNA5 ISA.

__device__ __forceinline__ void tdm_load_w_tile(const half_t* gtile,
                                                unsigned lds_byte_addr) {
  unsigned long long ga = (unsigned long long)(size_t)gtile;
  u32x4 g0;
  g0[0] = 1u;                                   // count=1 (valid), user mode
  g0[1] = lds_byte_addr;                        // lds_addr
  g0[2] = (unsigned)(ga & 0xFFFFFFFFu);         // global_addr[31:0]
  g0[3] = (unsigned)((ga >> 32) & 0x1FFFFFFu)   // global_addr[56:32]
        | (2u << 30);                           // type = 2 (image)
  i32x8 g1;
  g1[0] = (1 << 16)     // data_size = 2 bytes
        | (1 << 20)     // pad_enable
        | (4 << 22)     // pad_interval: 32 dwords (128B) between pads
        | (3 << 25);    // pad_amount: 4 dwords (16B)
  g1[1] = (int)(512u << 16);   // tensor_dim0[15:0] at bits[63:48]
  g1[2] = (int)(512u << 16);   // tensor_dim1[15:0] at bits[95:80]
  g1[3] = (int)(64u << 16);    // tile_dim0 = 64 at bits[127:112]
  g1[4] = 32;                  // tile_dim1 = 32 at bits[143:128]
  g1[5] = 512;                 // tensor_dim0_stride[31:0] (elements)
  g1[6] = 0;
  g1[7] = 0;
  i32x4 z4 = {0, 0, 0, 0};
#if defined(__clang_major__) && (__clang_major__ >= 23)
  i32x8 z8 = {0, 0, 0, 0, 0, 0, 0, 0};
  __builtin_amdgcn_tensor_load_to_lds(g0, g1, z4, z4, z8, 0);
#else
  __builtin_amdgcn_tensor_load_to_lds(g0, g1, z4, z4, 0);
#endif
}

// ---------------- small utility kernels ----------------

__global__ __launch_bounds__(256) void k_cvt(const float* __restrict__ src,
                                             half_t* __restrict__ dst, int n4) {
  int i = blockIdx.x * 256 + threadIdx.x;
  if (i < n4) {
    float4 f = ((const float4*)src)[i];
    ((v4h*)dst)[i] = cvt4(f);
  }
}

__global__ __launch_bounds__(256) void k_copy(const float* __restrict__ src,
                                              float* __restrict__ dst, size_t n4) {
  size_t i = (size_t)blockIdx.x * 256 + threadIdx.x;
  if (i < n4) ((float4*)dst)[i] = ((const float4*)src)[i];
}

// ---------------- GEMM: C[M,512] = A[M,512](f32) @ W[512,512](f16) (+bias) ----
// 256 thr / 8 waves, tile 128x64. W tile double-buffered in LDS via TDM;
// A tile software-pipelined through registers (float4 loads, v4h LDS stores).

__global__ __launch_bounds__(256) void k_gemm(const float* __restrict__ A,
                                              const half_t* __restrict__ W,
                                              const float* __restrict__ bias,
                                              float* __restrict__ C) {
  __shared__ half_t As[128][40];      // 32 K + pad (80B rows, 16B aligned)
  __shared__ half_t Ws[2][32][72];    // 64 N + pad (144B rows, 16B aligned)
  int tid  = threadIdx.x;
  int lane = tid & 31;
  int wave = tid >> 5;
  int row0 = blockIdx.x * 128;
  int col0 = blockIdx.y * 64;

  v8f acc[4] = {v8f{}, v8f{}, v8f{}, v8f{}};

  // A staging map: 1024 float4s (128 rows x 8), 4 per thread
  int f4r[4], f4c[4];
  float4 areg[4];
#pragma unroll
  for (int q = 0; q < 4; ++q) {
    int f4 = tid + 256 * q;
    f4r[q] = f4 >> 3;
    f4c[q] = (f4 & 7) << 2;
  }
  const float4* Af = (const float4*)A;

  // prologue: first A tile into registers, first W tile via TDM
#pragma unroll
  for (int q = 0; q < 4; ++q)
    areg[q] = Af[(size_t)(row0 + f4r[q]) * (Dd / 4) + (f4c[q] >> 2)];
  if (tid < 32)
    tdm_load_w_tile(W + (size_t)col0, (unsigned)(size_t)&Ws[0][0][0]);

  for (int kt = 0; kt < Dd / 32; ++kt) {
    int kbuf = kt & 1;
    // store pipelined A regs -> LDS (packed 8B stores)
#pragma unroll
    for (int q = 0; q < 4; ++q)
      *(v4h*)&As[f4r[q]][f4c[q]] = cvt4(areg[q]);
    // issue next A tile global loads (latency hidden under compute)
    if (kt + 1 < Dd / 32) {
      int k0n = (kt + 1) * 32;
#pragma unroll
      for (int q = 0; q < 4; ++q)
        areg[q] = Af[(size_t)(row0 + f4r[q]) * (Dd / 4) + ((k0n + f4c[q]) >> 2)];
    }
    if (tid < 32) __builtin_amdgcn_s_wait_tensorcnt(0);  // current W tile landed
    __syncthreads();  // As visible; everyone past previous tail barrier
    // safe to overwrite the other W buffer now: issue next TDM, overlaps compute
    if (tid < 32 && kt + 1 < Dd / 32)
      tdm_load_w_tile(W + (size_t)((kt + 1) * 32) * Dd + col0,
                      (unsigned)(size_t)&Ws[kbuf ^ 1][0][0]);

    v16h a = frag_a_lds(&As[0][0], 40, wave * 16, 0, lane);
#pragma unroll
    for (int c = 0; c < 4; ++c) {
      v16h b;  // B frag: lane%16 = N col, halves = K rows (table order)
#pragma unroll
      for (int j = 0; j < 16; ++j) b[j] = Ws[kbuf][((lane >> 4) << 3) + (j & 7) + ((j >> 3) << 4)][c * 16 + (lane & 15)];
      acc[c] = wmma_f16(a, b, acc[c]);
    }
    __syncthreads();  // done reading As before next overwrite
  }

#pragma unroll
  for (int c = 0; c < 4; ++c) {
    int n  = col0 + c * 16 + (lane & 15);
    float bv = bias ? bias[n] : 0.f;
    int mb = row0 + wave * 16 + ((lane >> 4) << 3);
#pragma unroll
    for (int i = 0; i < 8; ++i)
      C[(size_t)(mb + i) * Dd + n] = acc[c][i] + bv;
  }
}

// ---------------- GAT attention: one block per graph ----------------
// out = elu(h_in + mask_i * (softmax(leaky(f1_i+f2_j)) @ Wh))

__global__ __launch_bounds__(256) void k_gat(const float* __restrict__ Wh,
                                             const float* __restrict__ h_in,
                                             const float* __restrict__ avec,
                                             const unsigned char* __restrict__ mask,
                                             float* __restrict__ out) {
  int m = blockIdx.x;  // graph id
  __shared__ half_t WhS[Nn][520];          // 32x512 f16, padded (1040B rows)
  __shared__ half_t attnS[Hh][Nn][36];     // per-head 32x32 f16, padded
  __shared__ float  f1s[Hh][Nn], f2s[Hh][Nn];
  __shared__ float  av[2 * HD];
  __shared__ unsigned char mk[Nn];

  int tid = threadIdx.x, lane = tid & 31, wave = tid >> 5;
  av[tid] = avec[tid];
  if (tid < Nn) mk[tid] = mask[(size_t)m * Nn + tid];
  const float4* Whf = (const float4*)(Wh + (size_t)m * Nn * Dd);
#pragma unroll
  for (int i = tid; i < Nn * Dd / 4; i += 256) {   // 4096 float4s
    int r = i >> 7, c4 = (i & 127) << 2;
    *(v4h*)&WhS[r][c4] = cvt4(Whf[i]);
  }
  __syncthreads();

  if (tid < Hh * Nn) {
    int hh = tid >> 5, nn = tid & 31;
    float s1 = 0.f, s2 = 0.f;
#pragma unroll 8
    for (int d = 0; d < HD; ++d) {
      float w = (float)WhS[nn][hh * HD + d];
      s1 += w * av[d];
      s2 += w * av[HD + d];
    }
    f1s[hh][nn] = s1;
    f2s[hh][nn] = s2;
  }
  __syncthreads();

  if (tid < Hh * Nn) {  // masked softmax over j for row (hh, ii)
    int hh = tid >> 5, ii = tid & 31;
    bool mi = mk[ii] != 0;
    float e[Nn];
    float mx = -3.4e38f;
#pragma unroll
    for (int j = 0; j < Nn; ++j) {
      float v;
      if (mi && mk[j]) {
        v = f1s[hh][ii] + f2s[hh][j];
        v = v > 0.f ? v : 0.2f * v;  // leaky_relu
      } else {
        v = -3.4e38f;
      }
      e[j] = v;
      mx = v > mx ? v : mx;
    }
    float sum = 0.f;
#pragma unroll
    for (int j = 0; j < Nn; ++j) {
      float t = (e[j] > -1e38f) ? __expf(e[j] - mx) : 0.f;
      e[j] = t;
      sum += t;
    }
    float inv = sum > 0.f ? 1.f / sum : 0.f;
#pragma unroll
    for (int j = 0; j < Nn; ++j) attnS[hh][ii][j] = (half_t)(e[j] * inv);
  }
  __syncthreads();

  // hp = attn @ Wh (per head 32x32 @ 32x128): 64 tiles of 16x16, 8 per wave
#pragma unroll
  for (int tt = 0; tt < 8; ++tt) {
    int tile = wave * 8 + tt;
    int hh = tile >> 4, rem = tile & 15;
    int ri = rem >> 3, ci = rem & 7;
    v16h a = frag_a_lds(&attnS[hh][0][0], 36, ri * 16, 0, lane);
    v16h b;  // B frag: lane%16 = feature col, halves = node j (K, table order)
#pragma unroll
    for (int j = 0; j < 16; ++j) b[j] = WhS[((lane >> 4) << 3) + (j & 7) + ((j >> 3) << 4)][hh * HD + ci * 16 + (lane & 15)];
    v8f acc = {};
    acc = wmma_f16(a, b, acc);

    int nn = hh * HD + ci * 16 + (lane & 15);
    int mb = ri * 16 + ((lane >> 4) << 3);
#pragma unroll
    for (int i2 = 0; i2 < 8; ++i2) {
      int node = mb + i2;
      float hp = acc[i2] * (mk[node] ? 1.f : 0.f);
      size_t idx = ((size_t)m * Nn + node) * Dd + nn;
      float v = h_in[idx] + hp;
      out[idx] = v > 0.f ? v : (__expf(v) - 1.f);  // elu
    }
  }
}

// ---------------- temporal attention: scores + softmax -> P (f16) ----------

__global__ __launch_bounds__(256) void k_scores(const float* __restrict__ Q,
                                                const float* __restrict__ K,
                                                const unsigned char* __restrict__ mask,
                                                half_t* __restrict__ P) {
  int h = blockIdx.x & 3;
  int bn = blockIdx.x >> 2;
  int b = bn >> 5, n = bn & 31;
  __shared__ half_t Qs[Tt][136];
  __shared__ half_t Ks[Tt][136];
  __shared__ float  Ss[Tt][68];
  __shared__ unsigned char mk[Tt];
  int tid = threadIdx.x, lane = tid & 31, wave = tid >> 5;

#pragma unroll
  for (int i = tid; i < Tt * HD / 4; i += 256) {   // 2048 float4s
    int t = i >> 5, d4 = (i & 31) << 2;
    size_t row = (size_t)(b * Tt + t) * Nn + n;
    *(v4h*)&Qs[t][d4] = cvt4(*(const float4*)&Q[row * Dd + h * HD + d4]);
    *(v4h*)&Ks[t][d4] = cvt4(*(const float4*)&K[row * Dd + h * HD + d4]);
  }
  if (tid < Tt) mk[tid] = mask[(size_t)(b * Tt + tid) * Nn + n];
  __syncthreads();

  // 64x64 scores: 16 tiles, 2 per wave; K-dim = HD = 4 x 32
#pragma unroll
  for (int tt = 0; tt < 2; ++tt) {
    int tile = wave * 2 + tt;
    int qi = tile >> 2, ki = tile & 3;
    v8f acc = {};
#pragma unroll
    for (int k0 = 0; k0 < HD; k0 += 32) {
      v16h a = frag_a_lds(&Qs[0][0], 136, qi * 16, k0, lane);
      v16h bfr;  // B = K^T: lane%16 = key index, halves = d (K-dim, table order)
      int kk = k0 + ((lane >> 4) << 3);
#pragma unroll
      for (int j = 0; j < 16; ++j) bfr[j] = Ks[ki * 16 + (lane & 15)][kk + (j & 7) + ((j >> 3) << 4)];
      acc = wmma_f16(a, bfr, acc);
    }
    int nn = ki * 16 + (lane & 15);
    int mb = qi * 16 + ((lane >> 4) << 3);
#pragma unroll
    for (int i2 = 0; i2 < 8; ++i2)
      Ss[mb + i2][nn] = acc[i2] * 0.08838834764831845f;  // 1/sqrt(128)
  }
  __syncthreads();

  if (tid < Tt) {  // masked softmax over keys, one row per thread
    float mx = -3.4e38f;
#pragma unroll 8
    for (int j = 0; j < Tt; ++j) {
      float v = mk[j] ? Ss[tid][j] : -3.4e38f;
      mx = v > mx ? v : mx;
    }
    float sum = 0.f;
#pragma unroll 8
    for (int j = 0; j < Tt; ++j) {
      float e = mk[j] ? __expf(Ss[tid][j] - mx) : 0.f;
      Ss[tid][j] = e;
      sum += e;
    }
    float inv = sum > 0.f ? 1.f / sum : 0.f;
    size_t base = (size_t)blockIdx.x * (Tt * Tt) + (size_t)tid * Tt;
#pragma unroll 8
    for (int j = 0; j < Tt; ++j) P[base + j] = (half_t)(Ss[tid][j] * inv);
  }
}

// ---------------- temporal attention: O = P @ V ----------------

__global__ __launch_bounds__(256) void k_av(const half_t* __restrict__ P,
                                            const float* __restrict__ V,
                                            float* __restrict__ O) {
  int h = blockIdx.x & 3;
  int bn = blockIdx.x >> 2;
  int b = bn >> 5, n = bn & 31;
  __shared__ half_t Ps[Tt][72];
  __shared__ half_t Vs[Tt][136];
  int tid = threadIdx.x, lane = tid & 31, wave = tid >> 5;

  const v4h* Pv = (const v4h*)(P + (size_t)blockIdx.x * (Tt * Tt));
#pragma unroll
  for (int i = tid; i < Tt * Tt / 4; i += 256) {   // 1024 v4h
    int r = i >> 4, c4 = (i & 15) << 2;
    *(v4h*)&Ps[r][c4] = Pv[i];
  }
#pragma unroll
  for (int i = tid; i < Tt * HD / 4; i += 256) {
    int t = i >> 5, d4 = (i & 31) << 2;
    size_t row = (size_t)(b * Tt + t) * Nn + n;
    *(v4h*)&Vs[t][d4] = cvt4(*(const float4*)&V[row * Dd + h * HD + d4]);
  }
  __syncthreads();

  // O 64x128: 32 tiles, 4 per wave; K-dim = keys 64 = 2 x 32
#pragma unroll
  for (int tt = 0; tt < 4; ++tt) {
    int tile = wave * 4 + tt;
    int qi = tile >> 3, ci = tile & 7;
    v8f acc = {};
#pragma unroll
    for (int k0 = 0; k0 < Tt; k0 += 32) {
      v16h a = frag_a_lds(&Ps[0][0], 72, qi * 16, k0, lane);
      v16h bfr;  // B = V: lane%16 = feature col, halves = key (K, table order)
      int kk = k0 + ((lane >> 4) << 3);
#pragma unroll
      for (int j = 0; j < 16; ++j) bfr[j] = Vs[kk + (j & 7) + ((j >> 3) << 4)][ci * 16 + (lane & 15)];
      acc = wmma_f16(a, bfr, acc);
    }
    int nn = ci * 16 + (lane & 15);
    int mb = qi * 16 + ((lane >> 4) << 3);
#pragma unroll
    for (int i2 = 0; i2 < 8; ++i2) {
      int t = mb + i2;
      O[((size_t)(b * Tt + t) * Nn + n) * Dd + h * HD + nn] = acc[i2];
    }
  }
}

// ---------------- fused residual + layernorm (+mask) ----------------

__global__ __launch_bounds__(256) void k_ln(const float* __restrict__ X,
                                            const float* __restrict__ R,
                                            const float* __restrict__ g,
                                            const float* __restrict__ bta,
                                            const unsigned char* __restrict__ mask,
                                            float* __restrict__ out) {
  int row = blockIdx.x;
  int tid = threadIdx.x;
  size_t base = (size_t)row * Dd;
  float v0 = X[base + tid];
  float v1 = X[base + 256 + tid];
  if (R) {
    v0 += R[base + tid];
    v1 += R[base + 256 + tid];
  }
  __shared__ float sh1[256], sh2[256];
  sh1[tid] = v0 + v1;
  sh2[tid] = v0 * v0 + v1 * v1;
  __syncthreads();
  for (int s = 128; s > 0; s >>= 1) {
    if (tid < s) {
      sh1[tid] += sh1[tid + s];
      sh2[tid] += sh2[tid + s];
    }
    __syncthreads();
  }
  float mean = sh1[0] * (1.f / Dd);
  float var  = sh2[0] * (1.f / Dd) - mean * mean;
  float inv  = rsqrtf(var + 1e-5f);
  float mf   = mask ? (mask[row] ? 1.f : 0.f) : 1.f;
  out[base + tid]       = ((v0 - mean) * inv * g[tid] + bta[tid]) * mf;
  out[base + 256 + tid] = ((v1 - mean) * inv * g[tid + 256] + bta[tid + 256]) * mf;
}

// ---------------- host orchestration ----------------

struct LayerP {
  const float *Wg0, *ag0, *Wg1, *ag1, *gg, *gb, *Wk, *bk, *n1g, *n1b,
              *n2g, *n2b, *Wo, *bo, *Wq, *bq, *tg, *tb, *Wv, *bv;
};

extern "C" void kernel_launch(void* const* d_in, const int* in_sizes, int n_in,
                              void* d_out, int out_size, void* d_ws, size_t ws_size,
                              hipStream_t stream) {
  (void)in_sizes; (void)n_in; (void)out_size; (void)ws_size;
  const float* x_in = (const float*)d_in[0];
  const unsigned char* mask = (const unsigned char*)d_in[1];

  const size_t TOT = (size_t)MROWS * Dd;      // 16,777,216 elems
  const size_t SZB = TOT * sizeof(float);     // 64 MB
  char* wsb = (char*)d_ws;
  float* S[6];
  for (int i = 0; i < 6; ++i) S[i] = (float*)(wsb + (size_t)i * SZB);
  half_t* WF   = (half_t*)(wsb + 6 * SZB);                      // 18 x 262144 f16
  half_t* Pbuf = (half_t*)(wsb + 6 * SZB + (size_t)18 * Dd * Dd * sizeof(half_t));

  // Param leaves: pytree order (sorted dict keys per layer):
  // gat[W0,a0,W1,a1], gnn_ln[g,b], k[W,b], n1[g,b], n2[g,b], o[W,b], q[W,b],
  // tln[g,b], v[W,b]  -> 20 leaves/layer after the 2 positional inputs.
  LayerP Lp[3];
  int idx = 2;
  for (int li = 0; li < 3; ++li) {
    const float** p = (const float**)&Lp[li];
    for (int j = 0; j < 20; ++j) p[j] = (const float*)d_in[idx + j];
    idx += 20;
  }

  // Pre-convert weight matrices to f16: slots per layer {g0,g1,q,k,v,o}
  const int WELEMS = Dd * Dd;
  for (int li = 0; li < 3; ++li) {
    const float* srcs[6] = {Lp[li].Wg0, Lp[li].Wg1, Lp[li].Wq,
                            Lp[li].Wk,  Lp[li].Wv,  Lp[li].Wo};
    for (int s = 0; s < 6; ++s)
      k_cvt<<<WELEMS / 1024, 256, 0, stream>>>(srcs[s],
                                               WF + ((size_t)li * 6 + s) * WELEMS,
                                               WELEMS / 4);
  }

  k_copy<<<(int)(TOT / 1024), 256, 0, stream>>>(x_in, S[0], TOT / 4);

  dim3 ggrid(MROWS / 128, Dd / 64);  // (256, 8)
  for (int li = 0; li < 3; ++li) {
    const LayerP& P = Lp[li];
    half_t* Wg0 = WF + ((size_t)li * 6 + 0) * WELEMS;
    half_t* Wg1 = WF + ((size_t)li * 6 + 1) * WELEMS;
    half_t* Wq  = WF + ((size_t)li * 6 + 2) * WELEMS;
    half_t* Wk  = WF + ((size_t)li * 6 + 3) * WELEMS;
    half_t* Wv  = WF + ((size_t)li * 6 + 4) * WELEMS;
    half_t* Wo  = WF + ((size_t)li * 6 + 5) * WELEMS;

    // ---- GNN: h = x; h = elu(h + gat(h)) x2 ----
    k_copy<<<(int)(TOT / 1024), 256, 0, stream>>>(S[0], S[2], TOT / 4);
    float* h = S[2];
    float* ho = S[4];
    for (int gi = 0; gi < 2; ++gi) {
      k_gemm<<<ggrid, 256, 0, stream>>>(h, gi ? Wg1 : Wg0, nullptr, S[3]);
      k_gat<<<NGRAPH, 256, 0, stream>>>(S[3], h, gi ? P.ag1 : P.ag0, mask, ho);
      float* t = h; h = ho; ho = t;
    }
    // h == S[2] again after two swaps
    k_ln<<<MROWS, 256, 0, stream>>>(h, nullptr, P.gg, P.gb, mask, S[1]);      // y=LN(h)*m
    k_ln<<<MROWS, 256, 0, stream>>>(S[1], S[0], P.n1g, P.n1b, nullptr, S[1]); // y=LN(y+x)

    // ---- temporal MHA ----
    k_gemm<<<ggrid, 256, 0, stream>>>(S[1], Wq, P.bq, S[3]);
    k_gemm<<<ggrid, 256, 0, stream>>>(S[1], Wk, P.bk, S[4]);
    k_gemm<<<ggrid, 256, 0, stream>>>(S[1], Wv, P.bv, S[5]);
    k_scores<<<Bb * Nn * Hh, 256, 0, stream>>>(S[3], S[4], mask, Pbuf);
    k_av<<<Bb * Nn * Hh, 256, 0, stream>>>(Pbuf, S[5], S[2]);
    k_gemm<<<ggrid, 256, 0, stream>>>(S[2], Wo, P.bo, S[3]);
    k_ln<<<MROWS, 256, 0, stream>>>(S[3], S[1], P.tg, P.tb, mask, S[3]);      // o=LN(o+y)*m
    float* outp = (li == 2) ? (float*)d_out : S[0];
    k_ln<<<MROWS, 256, 0, stream>>>(S[3], S[0], P.n2g, P.n2b, nullptr, outp); // x=LN(o+x)
  }
}
